// BayesLinearLayer_17824114278627
// MI455X (gfx1250) — matrix operand order repaired and checked by reading, not compiled, not run on hardware
//
#include <hip/hip_runtime.h>
#include <math.h>

#define N_ROWS  8192
#define IN_DIM  1024
#define OUT_DIM 1024

typedef __attribute__((ext_vector_type(16))) _Float16 v16h;
typedef __attribute__((ext_vector_type(8)))  _Float16 v8h;
typedef __attribute__((ext_vector_type(4)))  _Float16 v4h;
typedef __attribute__((ext_vector_type(8)))  float    v8f;

// ---------------------------------------------------------------------------
// Prep: convert x (f32) -> f16
// ---------------------------------------------------------------------------
__global__ __launch_bounds__(256) void conv_x_kernel(const float* __restrict__ x,
                                                     _Float16* __restrict__ xh,
                                                     int total4) {
    int i = blockIdx.x * blockDim.x + threadIdx.x;
    if (i >= total4) return;
    float4 v = ((const float4*)x)[i];
    v4h h;
    h[0] = (_Float16)v.x; h[1] = (_Float16)v.y;
    h[2] = (_Float16)v.z; h[3] = (_Float16)v.w;
    ((v4h*)xh)[i] = h;
}

// ---------------------------------------------------------------------------
// Prep: W_mean -> f16 ; softplus(W_pho)^2 -> f16
// ---------------------------------------------------------------------------
__global__ __launch_bounds__(256) void conv_w_kernel(const float* __restrict__ wm,
                                                     const float* __restrict__ wp,
                                                     _Float16* __restrict__ wmh,
                                                     _Float16* __restrict__ wvh,
                                                     int total4) {
    int i = blockIdx.x * blockDim.x + threadIdx.x;
    if (i >= total4) return;
    float4 m = ((const float4*)wm)[i];
    float4 p = ((const float4*)wp)[i];
    float mv[4] = {m.x, m.y, m.z, m.w};
    float pv[4] = {p.x, p.y, p.z, p.w};
    v4h hm, hv;
#pragma unroll
    for (int j = 0; j < 4; ++j) {
        float pj = pv[j];
        float sp = (pj > 20.0f) ? pj
                 : (pj < -20.0f) ? expf(pj)
                 : log1pf(expf(pj));
        hm[j] = (_Float16)mv[j];
        hv[j] = (_Float16)(sp * sp);
    }
    ((v4h*)wmh)[i] = hm;
    ((v4h*)wvh)[i] = hv;
}

// ---------------------------------------------------------------------------
// Stateless per-element normal (deterministic; Box-Muller over hash)
// ---------------------------------------------------------------------------
__device__ __forceinline__ unsigned hash_u32(unsigned x) {
    x ^= x >> 16; x *= 0x7feb352du;
    x ^= x >> 15; x *= 0x846ca68bu;
    x ^= x >> 16;
    return x;
}

__device__ __forceinline__ float normal_at(unsigned idx) {
    unsigned r1 = hash_u32(idx ^ 0x9e3779b9u);
    unsigned r2 = hash_u32(idx + 0x85ebca6bu);
    float u1 = ((float)r1 + 1.0f) * (1.0f / 4294967296.0f);  // (0,1]
    float u2 = (float)r2 * (1.0f / 4294967296.0f);
    float r  = sqrtf(-2.0f * logf(u1));
    return r * cosf(6.28318530718f * u2);
}

// ---------------------------------------------------------------------------
// Fragment loaders: contiguous k order (k = 16*(lane/16) + s), shared by A and B
// A 16x32 f16: lanes 0-15 -> K {0..15}; lanes 16-31 -> K {16..31}
//   per-lane: two contiguous 8-half (16 B) chunks at +ahalf and +8+ahalf.
// B 32x16 f16: lane = column, contiguous 16 halfs (32 B) at +bhalf.
// ---------------------------------------------------------------------------
__device__ __forceinline__ v16h load_frag_a(const _Float16* p) {
    v8h lo = *(const v8h*)(p);
    v8h hi = *(const v8h*)(p + 8);
    return __builtin_shufflevector(lo, hi, 0,1,2,3,4,5,6,7,8,9,10,11,12,13,14,15);
}

// ---------------------------------------------------------------------------
// Fused dual-GEMM: gamma = x @ Wm^T , delta = x^2 @ Wv^T, then reparam sample.
// Block = 256 thr (8 waves). Wave tile = 32x32 output, both gamma+delta.
// Block tile = 128 (M) x 64 (N).
// ---------------------------------------------------------------------------
__global__ __launch_bounds__(256) void bayes_gemm_kernel(
    const _Float16* __restrict__ xh,
    const _Float16* __restrict__ wmh,
    const _Float16* __restrict__ wvh,
    float* __restrict__ out) {

    const int lane = threadIdx.x & 31;
    const int wave = threadIdx.x >> 5;
    const int wm_i = wave & 3;   // 4 waves along M
    const int wn_i = wave >> 2;  // 2 waves along N

    const int mblk = blockIdx.y * 128 + wm_i * 32;
    const int nblk = blockIdx.x * 64  + wn_i * 32;

    const int l16   = lane & 15;
    const int ahalf = (lane < 16) ? 0 : 16;   // A-fragment K sub-offset
    const int bhalf = (lane < 16) ? 0 : 16;   // B-fragment K sub-offset

    const _Float16* a0p = xh  + (size_t)(mblk      + l16) * IN_DIM + ahalf;
    const _Float16* a1p = xh  + (size_t)(mblk + 16 + l16) * IN_DIM + ahalf;
    const _Float16* bm0p = wmh + (size_t)(nblk      + l16) * IN_DIM + bhalf;
    const _Float16* bm1p = wmh + (size_t)(nblk + 16 + l16) * IN_DIM + bhalf;
    const _Float16* bv0p = wvh + (size_t)(nblk      + l16) * IN_DIM + bhalf;
    const _Float16* bv1p = wvh + (size_t)(nblk + 16 + l16) * IN_DIM + bhalf;

    v8f accG[2][2] = {};
    v8f accD[2][2] = {};

#pragma unroll 2
    for (int kb = 0; kb < IN_DIM; kb += 32) {
        if (kb + 32 < IN_DIM) {  // prefetch next K block (global_prefetch_b8)
            __builtin_prefetch(a0p + kb + 32, 0, 1);
            __builtin_prefetch(a1p + kb + 32, 0, 1);
            __builtin_prefetch(bm0p + kb + 32, 0, 1);
            __builtin_prefetch(bv0p + kb + 32, 0, 1);
        }

        v16h a0 = load_frag_a(a0p + kb);
        v16h a1 = load_frag_a(a1p + kb);
        v16h a0s = a0 * a0;  // x^2 in-register (v_pk_mul_f16)
        v16h a1s = a1 * a1;

        v16h bm0 = *(const v16h*)(bm0p + kb);
        v16h bm1 = *(const v16h*)(bm1p + kb);
        v16h bv0 = *(const v16h*)(bv0p + kb);
        v16h bv1 = *(const v16h*)(bv1p + kb);

        accG[0][0] = __builtin_amdgcn_wmma_f32_16x16x32_f16(false, a0,  false, bm0, (short)0, accG[0][0], false, false);
        accG[0][1] = __builtin_amdgcn_wmma_f32_16x16x32_f16(false, a0,  false, bm1, (short)0, accG[0][1], false, false);
        accG[1][0] = __builtin_amdgcn_wmma_f32_16x16x32_f16(false, a1,  false, bm0, (short)0, accG[1][0], false, false);
        accG[1][1] = __builtin_amdgcn_wmma_f32_16x16x32_f16(false, a1,  false, bm1, (short)0, accG[1][1], false, false);
        accD[0][0] = __builtin_amdgcn_wmma_f32_16x16x32_f16(false, a0s, false, bv0, (short)0, accD[0][0], false, false);
        accD[0][1] = __builtin_amdgcn_wmma_f32_16x16x32_f16(false, a0s, false, bv1, (short)0, accD[0][1], false, false);
        accD[1][0] = __builtin_amdgcn_wmma_f32_16x16x32_f16(false, a1s, false, bv0, (short)0, accD[1][0], false, false);
        accD[1][1] = __builtin_amdgcn_wmma_f32_16x16x32_f16(false, a1s, false, bv1, (short)0, accD[1][1], false, false);
    }

    // Epilogue: C/D layout — lane = N col (mod 16), VGPR v = M row (+8 for hi lanes)
    const int rbase = (lane < 16) ? 0 : 8;
#pragma unroll
    for (int ti = 0; ti < 2; ++ti) {
#pragma unroll
        for (int tj = 0; tj < 2; ++tj) {
            const int o     = nblk + tj * 16 + l16;
            const int mbase = mblk + ti * 16 + rbase;
#pragma unroll
            for (int v = 0; v < 8; ++v) {
                const int n = mbase + v;
                float g = accG[ti][tj][v];
                float d = accD[ti][tj][v];
                float z = normal_at((unsigned)(n * OUT_DIM + o));
                out[(size_t)n * OUT_DIM + o] = g + sqrtf(fmaxf(d, 0.0f)) * z;
            }
        }
    }
}

// ---------------------------------------------------------------------------
// Launcher
// ---------------------------------------------------------------------------
extern "C" void kernel_launch(void* const* d_in, const int* in_sizes, int n_in,
                              void* d_out, int out_size, void* d_ws, size_t ws_size,
                              hipStream_t stream) {
    const float* x  = (const float*)d_in[0];
    const float* wm = (const float*)d_in[1];
    const float* wp = (const float*)d_in[2];
    float* out = (float*)d_out;

    // workspace layout: x_f16 (16 MB) | wmean_f16 (2 MB) | wvar_f16 (2 MB)
    _Float16* xh  = (_Float16*)d_ws;
    _Float16* wmh = xh  + (size_t)N_ROWS  * IN_DIM;
    _Float16* wvh = wmh + (size_t)OUT_DIM * IN_DIM;

    const int x4 = (N_ROWS * IN_DIM) / 4;      // 2,097,152 threads
    conv_x_kernel<<<x4 / 256, 256, 0, stream>>>(x, xh, x4);

    const int w4 = (OUT_DIM * IN_DIM) / 4;     // 262,144 threads
    conv_w_kernel<<<w4 / 256, 256, 0, stream>>>(wm, wp, wmh, wvh, w4);

    dim3 grid(OUT_DIM / 64, N_ROWS / 128);     // (16, 64)
    bayes_gemm_kernel<<<grid, 256, 0, stream>>>(xh, wmh, wvh, out);
}
